// GNBlock_32993938768001
// MI455X (gfx1250) — matrix operand order repaired and checked by reading, not compiled, not run on hardware
//
#include <hip/hip_runtime.h>
#include <hip/hip_bf16.h>

// ---------------- problem constants ----------------
#define Bc 4
#define Nn 8192
#define Ee 32768
#define Kc 16
#define Dd 256
#define EPSF 1e-3f

typedef __attribute__((ext_vector_type(16))) __bf16 bf16x16;
typedef __attribute__((ext_vector_type(8)))  __bf16 bf16x8;
typedef __attribute__((ext_vector_type(8)))  float  f32x8;

// ---------------------------------------------------------------------------
// Weight swizzle: W [Ktot,256] fp32 row-major -> bf16 fragments.
// Fragment order: frag[kt][nt][lane][i], lane L: n = nt*16 + (L&15),
// element i: k = kt*32 + (i>=8 ? 16 : 0) + (L>=16 ? 8 : 0) + (i&7)   (dense 16-bit B 32x16 layout).
// ---------------------------------------------------------------------------
__global__ __launch_bounds__(256) void swizzle_w_kernel(
    const float* __restrict__ W, __bf16* __restrict__ out, int kTiles) {
  int t = blockIdx.x * 256 + threadIdx.x;
  int total = kTiles * 16 * 32 * 16;
  if (t >= total) return;
  int i  = t & 15;
  int L  = (t >> 4) & 31;
  int nt = (t >> 9) & 15;
  int kt = t >> 13;
  int k  = kt * 32 + ((i >> 3) << 4) + ((L >> 4) << 3) + (i & 7);
  int n  = nt * 16 + (L & 15);
  out[t] = (__bf16)W[k * Dd + n];
}

__global__ __launch_bounds__(256) void zero_acc_kernel(float* accE, float* accV) {
  int t = blockIdx.x * 256 + threadIdx.x;
  if (t < Bc * Dd) accE[t] = 0.f;
  else if (t < 2 * Bc * Dd) accV[t - Bc * Dd] = 0.f;
}

// ---------------------------------------------------------------------------
// phi_edge: per block = 32 edges x 256 outputs. K = 1024 (32 k-tiles).
// Each wave: 2 n-tiles x 2 m-tiles -> each B fragment feeds 2 WMMAs and the
// full weight matrix is read once per 32 rows (halves L2 B-traffic vs 16).
// ---------------------------------------------------------------------------
#define ASTRIDE_E 1032   // 1024 + 8 bf16 pad -> conflict-free fragment reads
#define CSTRIDE   260

__global__ __launch_bounds__(256) void phi_edge_kernel(
    const float* __restrict__ vertex, const float* __restrict__ edges,
    const float* __restrict__ glob,   const int*   __restrict__ eidx,
    const __bf16* __restrict__ wB,    const float* __restrict__ bias,
    const float* __restrict__ lng,    const float* __restrict__ lnb,
    float* __restrict__ outE) {
  __shared__ __attribute__((aligned(32))) __bf16 xA[32 * ASTRIDE_E];
  __shared__ float ldsC[32 * CSTRIDE];
  __shared__ float sred[256], s2red[256];
  __shared__ float muS[32], rsS[32];

  const int tid   = threadIdx.x;
  const int b     = blockIdx.x / (Ee / 32);
  const int eBase = (blockIdx.x % (Ee / 32)) * 32;

  // ---- build concat rows [edges | vs | vr | global] as bf16 in LDS ----
  {
    const int r = tid >> 3, s = tid & 7;          // row 0..31, 128-col segment
    const int e = eBase + r;
    const int col0 = s * 128;
    const float* src;
    if (col0 < 256) {
      src = edges + ((size_t)b * Ee + e) * Dd + col0;
    } else if (col0 < 512) {
      int sn = eidx[((size_t)b * Ee + e) * 2 + 0];
      src = vertex + ((size_t)b * Nn + sn) * Dd + (col0 - 256);
    } else if (col0 < 768) {
      int rn = eidx[((size_t)b * Ee + e) * 2 + 1];
      src = vertex + ((size_t)b * Nn + rn) * Dd + (col0 - 512);
    } else {
      src = glob + (size_t)b * Dd + (col0 - 768);
    }
    __bf16* dst = &xA[r * ASTRIDE_E + col0];
    const float4* s4 = (const float4*)src;
    for (int j = 0; j < 32; ++j) {
      float4 v = s4[j];
      union { __bf16 h[4]; uint2 u; } p;
      p.h[0] = (__bf16)v.x; p.h[1] = (__bf16)v.y;
      p.h[2] = (__bf16)v.z; p.h[3] = (__bf16)v.w;
      *(uint2*)(dst + j * 4) = p.u;
    }
  }
  __syncthreads();

  // ---- WMMA GEMM: 8 waves x 2 n-tiles x 2 m-tiles, 32 k-tiles ----
  const int lane = tid & 31, wave = tid >> 5;
  const int h = lane >> 4, mrow = lane & 15;
  f32x8 c00 = {}; f32x8 c01 = {}; f32x8 c10 = {}; f32x8 c11 = {};
  const bf16x16* bp = (const bf16x16*)wB;
  for (int kt = 0; kt < 32; ++kt) {
    union { bf16x16 v; bf16x8 p[2]; } a0, a1;
    const __bf16* rp0 = &xA[mrow * ASTRIDE_E + kt * 32 + h * 8];
    const __bf16* rp1 = rp0 + 16 * ASTRIDE_E;
    a0.p[0] = *(const bf16x8*)rp0;
    a0.p[1] = *(const bf16x8*)(rp0 + 16);
    a1.p[0] = *(const bf16x8*)rp1;
    a1.p[1] = *(const bf16x8*)(rp1 + 16);
    bf16x16 b0 = bp[(kt * 16 + 2 * wave)     * 32 + lane];
    bf16x16 b1 = bp[(kt * 16 + 2 * wave + 1) * 32 + lane];
    c00 = __builtin_amdgcn_wmma_f32_16x16x32_bf16(false, a0.v, false, b0, (short)0, c00, false, false);
    c01 = __builtin_amdgcn_wmma_f32_16x16x32_bf16(false, a0.v, false, b1, (short)0, c01, false, false);
    c10 = __builtin_amdgcn_wmma_f32_16x16x32_bf16(false, a1.v, false, b0, (short)0, c10, false, false);
    c11 = __builtin_amdgcn_wmma_f32_16x16x32_bf16(false, a1.v, false, b1, (short)0, c11, false, false);
  }

  // ---- bias + relu -> LDS (C/D layout: lane half selects M 0-7 / 8-15) ----
  {
    int col0 = (2 * wave) * 16 + mrow, col1 = col0 + 16;
    float bb0 = bias[col0], bb1 = bias[col1];
    for (int j = 0; j < 8; ++j) {
      int row0 = j + 8 * h, row1 = row0 + 16;
      ldsC[row0 * CSTRIDE + col0] = fmaxf(c00[j] + bb0, 0.f);
      ldsC[row0 * CSTRIDE + col1] = fmaxf(c01[j] + bb1, 0.f);
      ldsC[row1 * CSTRIDE + col0] = fmaxf(c10[j] + bb0, 0.f);
      ldsC[row1 * CSTRIDE + col1] = fmaxf(c11[j] + bb1, 0.f);
    }
  }
  __syncthreads();

  // ---- residual + LayerNorm over D=256 ----
  const int r = tid >> 3, g = tid & 7;            // row, 32-col group
  const int e = eBase + r;
  const float* res = edges + ((size_t)b * Ee + e) * Dd + g * 32;
  float s1 = 0.f, s2 = 0.f;
  for (int cc = 0; cc < 32; ++cc) {
    float x = ldsC[r * CSTRIDE + g * 32 + cc] + res[cc];
    ldsC[r * CSTRIDE + g * 32 + cc] = x;
    s1 += x; s2 += x * x;
  }
  sred[tid] = s1; s2red[tid] = s2;
  __syncthreads();
  if (g == 0) {
    float a = 0.f, q2 = 0.f;
    for (int q = 0; q < 8; ++q) { a += sred[r * 8 + q]; q2 += s2red[r * 8 + q]; }
    float mu = a * (1.f / 256.f);
    float var = q2 * (1.f / 256.f) - mu * mu;
    muS[r] = mu; rsS[r] = rsqrtf(var + EPSF);
  }
  __syncthreads();
  float mu = muS[r], rs = rsS[r];
  float* op = outE + ((size_t)b * Ee + e) * Dd + g * 32;
  for (int cc = 0; cc < 32; ++cc) {
    int col = g * 32 + cc;
    op[cc] = (ldsC[r * CSTRIDE + col] - mu) * rs * lng[col] + lnb[col];
  }
}

// ---------------------------------------------------------------------------
// phi_node: per block = 32 nodes x 256 outputs. K = 768 (24 k-tiles).
// ---------------------------------------------------------------------------
#define ASTRIDE_N 776    // 768 + 8 bf16 pad

__global__ __launch_bounds__(256) void phi_node_kernel(
    const float* __restrict__ vertex, const float* __restrict__ edgesNew,
    const float* __restrict__ glob,   const int*   __restrict__ cidx,
    const int*   __restrict__ vlens,  const __bf16* __restrict__ wB,
    const float* __restrict__ bias,   const float* __restrict__ lng,
    const float* __restrict__ lnb,    float* __restrict__ outV) {
  __shared__ __attribute__((aligned(32))) __bf16 xA[32 * ASTRIDE_N];
  __shared__ float ldsC[32 * CSTRIDE];
  __shared__ float sred[256], s2red[256];
  __shared__ float muS[32], rsS[32];

  const int tid   = threadIdx.x;
  const int b     = blockIdx.x / (Nn / 32);
  const int nBase = (blockIdx.x % (Nn / 32)) * 32;

  // ---- phase A: vertex (cols 0-255) and global (cols 512-767) ----
  {
    const int r = tid >> 3, s = tid & 7;          // row, 64-col segment
    const int n = nBase + r;
    const float* src; int dcol;
    if (s < 4) { src = vertex + ((size_t)b * Nn + n) * Dd + s * 64; dcol = s * 64; }
    else       { src = glob + (size_t)b * Dd + (s - 4) * 64;        dcol = 512 + (s - 4) * 64; }
    __bf16* dst = &xA[r * ASTRIDE_N + dcol];
    const float4* s4 = (const float4*)src;
    for (int j = 0; j < 16; ++j) {
      float4 v = s4[j];
      union { __bf16 h[4]; uint2 u; } p;
      p.h[0] = (__bf16)v.x; p.h[1] = (__bf16)v.y;
      p.h[2] = (__bf16)v.z; p.h[3] = (__bf16)v.w;
      *(uint2*)(dst + j * 4) = p.u;
    }
  }
  // ---- phase B: conn_agg (cols 256-511): mean of K=16 gathered edge rows ----
  {
    const int r = tid >> 3, g = tid & 7;          // row, 32-col group
    const int n = nBase + r;
    const int* ci = cidx + ((size_t)b * Nn + n) * Kc;
    float acc[32];
    for (int q = 0; q < 32; ++q) acc[q] = 0.f;
    for (int k = 0; k < Kc; ++k) {
      const float* ep = edgesNew + ((size_t)b * Ee + ci[k]) * Dd + g * 32;
      for (int q = 0; q < 8; ++q) {
        float4 v = *(const float4*)(ep + q * 4);
        acc[q * 4 + 0] += v.x; acc[q * 4 + 1] += v.y;
        acc[q * 4 + 2] += v.z; acc[q * 4 + 3] += v.w;
      }
    }
    int vl = vlens[(size_t)b * Nn + n];
    float inv = 1.f / (vl == 0 ? 1.f : (float)vl);
    __bf16* dst = &xA[r * ASTRIDE_N + 256 + g * 32];
    for (int q = 0; q < 8; ++q) {
      union { __bf16 h[4]; uint2 u; } p;
      p.h[0] = (__bf16)(acc[q * 4 + 0] * inv); p.h[1] = (__bf16)(acc[q * 4 + 1] * inv);
      p.h[2] = (__bf16)(acc[q * 4 + 2] * inv); p.h[3] = (__bf16)(acc[q * 4 + 3] * inv);
      *(uint2*)(dst + q * 4) = p.u;
    }
  }
  __syncthreads();

  // ---- WMMA GEMM: 24 k-tiles ----
  const int lane = tid & 31, wave = tid >> 5;
  const int h = lane >> 4, mrow = lane & 15;
  f32x8 c00 = {}; f32x8 c01 = {}; f32x8 c10 = {}; f32x8 c11 = {};
  const bf16x16* bp = (const bf16x16*)wB;
  for (int kt = 0; kt < 24; ++kt) {
    union { bf16x16 v; bf16x8 p[2]; } a0, a1;
    const __bf16* rp0 = &xA[mrow * ASTRIDE_N + kt * 32 + h * 8];
    const __bf16* rp1 = rp0 + 16 * ASTRIDE_N;
    a0.p[0] = *(const bf16x8*)rp0;
    a0.p[1] = *(const bf16x8*)(rp0 + 16);
    a1.p[0] = *(const bf16x8*)rp1;
    a1.p[1] = *(const bf16x8*)(rp1 + 16);
    bf16x16 b0 = bp[(kt * 16 + 2 * wave)     * 32 + lane];
    bf16x16 b1 = bp[(kt * 16 + 2 * wave + 1) * 32 + lane];
    c00 = __builtin_amdgcn_wmma_f32_16x16x32_bf16(false, a0.v, false, b0, (short)0, c00, false, false);
    c01 = __builtin_amdgcn_wmma_f32_16x16x32_bf16(false, a0.v, false, b1, (short)0, c01, false, false);
    c10 = __builtin_amdgcn_wmma_f32_16x16x32_bf16(false, a1.v, false, b0, (short)0, c10, false, false);
    c11 = __builtin_amdgcn_wmma_f32_16x16x32_bf16(false, a1.v, false, b1, (short)0, c11, false, false);
  }

  {
    int col0 = (2 * wave) * 16 + mrow, col1 = col0 + 16;
    float bb0 = bias[col0], bb1 = bias[col1];
    for (int j = 0; j < 8; ++j) {
      int row0 = j + 8 * h, row1 = row0 + 16;
      ldsC[row0 * CSTRIDE + col0] = fmaxf(c00[j] + bb0, 0.f);
      ldsC[row0 * CSTRIDE + col1] = fmaxf(c01[j] + bb1, 0.f);
      ldsC[row1 * CSTRIDE + col0] = fmaxf(c10[j] + bb0, 0.f);
      ldsC[row1 * CSTRIDE + col1] = fmaxf(c11[j] + bb1, 0.f);
    }
  }
  __syncthreads();

  const int r = tid >> 3, g = tid & 7;
  const int n = nBase + r;
  const float* res = vertex + ((size_t)b * Nn + n) * Dd + g * 32;
  float s1 = 0.f, s2 = 0.f;
  for (int cc = 0; cc < 32; ++cc) {
    float x = ldsC[r * CSTRIDE + g * 32 + cc] + res[cc];
    ldsC[r * CSTRIDE + g * 32 + cc] = x;
    s1 += x; s2 += x * x;
  }
  sred[tid] = s1; s2red[tid] = s2;
  __syncthreads();
  if (g == 0) {
    float a = 0.f, q2 = 0.f;
    for (int q = 0; q < 8; ++q) { a += sred[r * 8 + q]; q2 += s2red[r * 8 + q]; }
    float mu = a * (1.f / 256.f);
    float var = q2 * (1.f / 256.f) - mu * mu;
    muS[r] = mu; rsS[r] = rsqrtf(var + EPSF);
  }
  __syncthreads();
  float mu = muS[r], rs = rsS[r];
  float* op = outV + ((size_t)b * Nn + n) * Dd + g * 32;
  for (int cc = 0; cc < 32; ++cc) {
    int col = g * 32 + cc;
    op[cc] = (ldsC[r * CSTRIDE + col] - mu) * rs * lng[col] + lnb[col];
  }
}

// ---------------------------------------------------------------------------
// Segmented row-sum of edges_new and vertex_new into fp32 accumulators.
// Blocks 0 .. B*64-1 : edges (512 rows each). Then B*16 blocks: vertices.
// ---------------------------------------------------------------------------
__global__ __launch_bounds__(256) void agg_kernel(
    const float* __restrict__ outE, const float* __restrict__ outV,
    float* __restrict__ accE, float* __restrict__ accV) {
  const int d = threadIdx.x;
  if (blockIdx.x < Bc * 64) {
    int b = blockIdx.x / 64, seg = blockIdx.x % 64;
    const float* p = outE + ((size_t)b * Ee + (size_t)seg * 512) * Dd;
    float s = 0.f;
    for (int r = 0; r < 512; ++r) s += p[(size_t)r * Dd + d];
    atomicAdd(&accE[b * Dd + d], s);
  } else {
    int idx = blockIdx.x - Bc * 64;
    int b = idx / 16, seg = idx % 16;
    const float* p = outV + ((size_t)b * Nn + (size_t)seg * 512) * Dd;
    float s = 0.f;
    for (int r = 0; r < 512; ++r) s += p[(size_t)r * Dd + d];
    atomicAdd(&accV[b * Dd + d], s);
  }
}

// ---------------------------------------------------------------------------
// phi_global: [B,768] x [768,256] scalar (tiny), relu + residual + BN.
// ---------------------------------------------------------------------------
__global__ __launch_bounds__(256) void phi_global_kernel(
    const float* __restrict__ glob, const float* __restrict__ accV,
    const float* __restrict__ accE, const int* __restrict__ vnodes,
    const int* __restrict__ vedges, const float* __restrict__ Wg,
    const float* __restrict__ bg,   const float* __restrict__ bng,
    const float* __restrict__ bnb,  const float* __restrict__ bnm,
    const float* __restrict__ bnv,  float* __restrict__ outG) {
  const int d = threadIdx.x;
  const float bn_scale = rsqrtf(bnv[d] + EPSF) * bng[d];
  for (int b = 0; b < Bc; ++b) {
    int vn = vnodes[b]; float invn = 1.f / (vn == 0 ? 1.f : (float)vn);
    int ve = vedges[b]; float inve = 1.f / (ve == 0 ? 1.f : (float)ve);
    float acc = bg[d];
    for (int f = 0; f < Dd; ++f)
      acc += glob[b * Dd + f] * Wg[f * Dd + d];
    for (int f = 0; f < Dd; ++f)
      acc += (accV[b * Dd + f] * invn) * Wg[(Dd + f) * Dd + d];
    for (int f = 0; f < Dd; ++f)
      acc += (accE[b * Dd + f] * inve) * Wg[(2 * Dd + f) * Dd + d];
    float resg = fmaxf(acc, 0.f) + glob[b * Dd + d];
    outG[b * Dd + d] = (resg - bnm[d]) * bn_scale + bnb[d];
  }
}

// ---------------------------------------------------------------------------
extern "C" void kernel_launch(void* const* d_in, const int* in_sizes, int n_in,
                              void* d_out, int out_size, void* d_ws, size_t ws_size,
                              hipStream_t stream) {
  const float* vertex = (const float*)d_in[0];
  const float* edges  = (const float*)d_in[1];
  const float* glob   = (const float*)d_in[2];
  const int*   eidx   = (const int*)d_in[3];
  const int*   cidx   = (const int*)d_in[4];
  const int*   vlens  = (const int*)d_in[5];
  const int*   vnodes = (const int*)d_in[6];
  const int*   vedges = (const int*)d_in[7];
  const float* W_e = (const float*)d_in[8];
  const float* b_e = (const float*)d_in[9];
  const float* ln_e_g = (const float*)d_in[10];
  const float* ln_e_b = (const float*)d_in[11];
  const float* W_n = (const float*)d_in[12];
  const float* b_n = (const float*)d_in[13];
  const float* ln_n_g = (const float*)d_in[14];
  const float* ln_n_b = (const float*)d_in[15];
  const float* W_g = (const float*)d_in[16];
  const float* b_g = (const float*)d_in[17];
  const float* bn_g = (const float*)d_in[18];
  const float* bn_b = (const float*)d_in[19];
  const float* bn_m = (const float*)d_in[20];
  const float* bn_v = (const float*)d_in[21];

  float* outE = (float*)d_out;                               // [B,E,D]
  float* outV = outE + (size_t)Bc * Ee * Dd;                 // [B,N,D]
  float* outG = outV + (size_t)Bc * Nn * Dd;                 // [B,D]

  char* ws = (char*)d_ws;
  __bf16* weB = (__bf16*)(ws);                                // 32*16*512 bf16 = 512KB
  __bf16* wnB = (__bf16*)(ws + 524288);                       // 24*16*512 bf16 = 384KB
  float*  accE = (float*)(ws + 917504);                       // B*256 fp32
  float*  accV = (float*)(ws + 921600);                       // B*256 fp32

  swizzle_w_kernel<<<1024, 256, 0, stream>>>(W_e, weB, 32);
  swizzle_w_kernel<<<768, 256, 0, stream>>>(W_n, wnB, 24);
  zero_acc_kernel<<<8, 256, 0, stream>>>(accE, accV);

  phi_edge_kernel<<<Bc * Ee / 32, 256, 0, stream>>>(
      vertex, edges, glob, eidx, weB, b_e, ln_e_g, ln_e_b, outE);

  phi_node_kernel<<<Bc * Nn / 32, 256, 0, stream>>>(
      vertex, outE, glob, cidx, vlens, wnB, b_n, ln_n_g, ln_n_b, outV);

  agg_kernel<<<Bc * 64 + Bc * 16, 256, 0, stream>>>(outE, outV, accE, accV);

  phi_global_kernel<<<1, 256, 0, stream>>>(
      glob, accV, accE, vnodes, vedges, W_g, b_g, bn_g, bn_b, bn_m, bn_v, outG);
}
